// GNNComponent_89610197663802
// MI455X (gfx1250) — matrix operand order repaired and checked by reading, not compiled, not run on hardware
//
#include <hip/hip_runtime.h>
#include <math.h>

// ---------------------------------------------------------------------------
// 3-layer GAT for MI455X (gfx1250, wave32).
//  - GEMMs: v_wmma_f32_16x16x32_f16; A via global b128 + v_cvt_pk_f16_f32,
//    W staged per block into LDS (f16, transposed) -> B = 2x ds_load_b128.
//    C tile staged through per-wave LDS -> coalesced global_store_b128.
//  - Edge softmax/aggregation: native f32 global atomics (relaxed, agent
//    scope). h/accum are 25.6 MB each -> fully L2-resident (192 MB L2).
// ---------------------------------------------------------------------------

typedef __attribute__((ext_vector_type(16))) _Float16 v16h;
typedef __attribute__((ext_vector_type(8)))  _Float16 v8h;
typedef __attribute__((ext_vector_type(8)))  float    v8f;

#define HID 64
#define NEG_SLOPE 0.2f
#define LDS_KP 136            // padded K stride (halves): supports K<=128, 16B aligned

__device__ __forceinline__ void atomicAddF(float* p, float v) {
  __hip_atomic_fetch_add(p, v, __ATOMIC_RELAXED, __HIP_MEMORY_SCOPE_AGENT);
}

// float atomic max via monotone int/uint mapping (m initialized to -inf).
__device__ __forceinline__ void atomicMaxF(float* p, float v) {
  if (v >= 0.0f)
    __hip_atomic_fetch_max((int*)p, __float_as_int(v),
                           __ATOMIC_RELAXED, __HIP_MEMORY_SCOPE_AGENT);
  else
    __hip_atomic_fetch_min((unsigned int*)p, __float_as_uint(v),
                           __ATOMIC_RELAXED, __HIP_MEMORY_SCOPE_AGENT);
}

// ---------------------------------------------------------------------------
// H[N,64] = A[N,K] @ W[K,64], K in {128, 64}.
// One wave per 16-row tile; 4 column tiles of 16; K-step 32.
// Fragment layouts per CDNA5 ISA 7.12.2:
//   A (16x32 f16): lane l<16 -> row=l, K = {0..7, 16..23}; lane>=16 -> K+8.
//   B (32x16 f16): lane l -> col=l&15, K = {0..7, 16..23}; lane>=16 -> K+8 (same as A).
//   C (16x16 f32): lane l, vgpr r -> row = r + 8*(l>>4), col = l&15.
// ---------------------------------------------------------------------------
__global__ __launch_bounds__(256)
void gat_gemm_wmma(const float* __restrict__ A, const float* __restrict__ W,
                   float* __restrict__ H, int N, int K) {
  __shared__ _Float16 Wt[HID * LDS_KP];      // f16 transposed weights
  __shared__ float    Ct[8][16 * HID];       // per-wave C staging (4 KB each)

  // ---- cooperative stage: W (f32, row-major [K][64]) -> Wt (f16, [64][KP])
  const int total = K * HID;
  for (int idx = threadIdx.x; idx < total; idx += 256) {
    const int k = idx >> 6;       // row of W
    const int n = idx & 63;       // col of W
    Wt[n * LDS_KP + k] = (_Float16)W[idx];
  }
  __syncthreads();

  const int lane = threadIdx.x & 31;
  const int rowTile = blockIdx.x * 8 + (threadIdx.x >> 5);
  if (rowTile * 16 >= N) return;          // after barrier: no divergence hazard

  const int half = lane >> 4;   // 0 or 1
  const int l16  = lane & 15;

  // clamped unconditional A row (invalid rows discarded at store)
  int arow = rowTile * 16 + l16;
  if (arow >= N) arow = N - 1;
  const float* Arow = A + (size_t)arow * K;

  v8f c[4];
  #pragma unroll
  for (int t = 0; t < 4; ++t) c[t] = (v8f){0.f,0.f,0.f,0.f,0.f,0.f,0.f,0.f};

  for (int kb = 0; kb < K; kb += 32) {
    // ---- A fragment: two 8-wide contiguous groups -> 4x global b128 ----
    const float4* Ap0 = (const float4*)(Arow + kb + half * 8);
    const float4* Ap1 = (const float4*)(Arow + kb + 16 + half * 8);
    const float4 a0 = Ap0[0], a1 = Ap0[1];
    const float4 a2 = Ap1[0], a3 = Ap1[1];
    v16h a;
    a[0]  = (_Float16)a0.x; a[1]  = (_Float16)a0.y;
    a[2]  = (_Float16)a0.z; a[3]  = (_Float16)a0.w;
    a[4]  = (_Float16)a1.x; a[5]  = (_Float16)a1.y;
    a[6]  = (_Float16)a1.z; a[7]  = (_Float16)a1.w;
    a[8]  = (_Float16)a2.x; a[9]  = (_Float16)a2.y;
    a[10] = (_Float16)a2.z; a[11] = (_Float16)a2.w;
    a[12] = (_Float16)a3.x; a[13] = (_Float16)a3.y;
    a[14] = (_Float16)a3.z; a[15] = (_Float16)a3.w;

    // ---- B fragments from LDS: 2x ds_load_b128 per column tile ----
    const int koff = kb + half * 8;
    #pragma unroll
    for (int t = 0; t < 4; ++t) {
      const _Float16* Bbase = &Wt[(t * 16 + l16) * LDS_KP + koff];
      const v8h b0 = *(const v8h*)(Bbase);
      const v8h b1 = *(const v8h*)(Bbase + 16);
      const v16h b = __builtin_shufflevector(b0, b1,
          0,1,2,3,4,5,6,7,8,9,10,11,12,13,14,15);
      c[t] = __builtin_amdgcn_wmma_f32_16x16x32_f16(
                 false, a, false, b, (short)0, c[t], false, false);
    }
  }

  // ---- store C: transpose via per-wave LDS, write coalesced b128 rows ----
  float* myC = Ct[threadIdx.x >> 5];
  #pragma unroll
  for (int t = 0; t < 4; ++t) {
    const int col = t * 16 + l16;
    #pragma unroll
    for (int r = 0; r < 8; ++r)
      myC[(half * 8 + r) * HID + col] = c[t][r];
  }
  // wave-local LDS ops are in-order; read back and store rows
  const int rowBase = rowTile * 16;
  const float4* myC4 = (const float4*)myC;
  if (rowBase + 16 <= N) {                 // fast path: full tile
    #pragma unroll
    for (int j = 0; j < 8; ++j) {
      const int flat = j * 32 + lane;      // 256 float4 slots in the tile
      const int r    = flat >> 4;
      const int c4   = flat & 15;
      ((float4*)(H + (size_t)(rowBase + r) * HID))[c4] = myC4[flat];
    }
  } else {                                 // ragged last tile
    #pragma unroll
    for (int j = 0; j < 8; ++j) {
      const int flat = j * 32 + lane;
      const int r    = flat >> 4;
      const int c4   = flat & 15;
      if (rowBase + r < N)
        ((float4*)(H + (size_t)(rowBase + r) * HID))[c4] = myC4[flat];
    }
  }
}

// ---------------------------------------------------------------------------
// Per-node: attention logits e_src/e_dst, init m=-inf, s=0, accum=0.
// ---------------------------------------------------------------------------
__global__ __launch_bounds__(256)
void node_prep(const float* __restrict__ H,
               const float* __restrict__ a_src, const float* __restrict__ a_dst,
               float* __restrict__ es, float* __restrict__ ed,
               float* __restrict__ m, float* __restrict__ s,
               float* __restrict__ accum, int N) {
  const int n = blockIdx.x * blockDim.x + threadIdx.x;
  if (n >= N) return;
  const float4* h4 = (const float4*)(H + (size_t)n * HID);
  const float4* as4 = (const float4*)a_src;
  const float4* ad4 = (const float4*)a_dst;
  float vs = 0.f, vd = 0.f;
  #pragma unroll
  for (int j = 0; j < HID / 4; ++j) {
    const float4 x = h4[j], sa = as4[j], da = ad4[j];
    vs = __builtin_fmaf(x.x, sa.x, vs); vd = __builtin_fmaf(x.x, da.x, vd);
    vs = __builtin_fmaf(x.y, sa.y, vs); vd = __builtin_fmaf(x.y, da.y, vd);
    vs = __builtin_fmaf(x.z, sa.z, vs); vd = __builtin_fmaf(x.z, da.z, vd);
    vs = __builtin_fmaf(x.w, sa.w, vs); vd = __builtin_fmaf(x.w, da.w, vd);
  }
  es[n] = vs;
  ed[n] = vd;
  m[n] = -__builtin_inff();
  s[n] = 0.f;
  float4* acc4 = (float4*)(accum + (size_t)n * HID);
  const float4 z = {0.f, 0.f, 0.f, 0.f};
  #pragma unroll
  for (int j = 0; j < HID / 4; ++j) acc4[j] = z;
}

// ---------------------------------------------------------------------------
// Pass 1: e = leaky_relu(e_src[src] + e_dst[dst]); store; segment max to m.
// Edges [0,E) from edge_index (int64); [E, E+N) are self loops.
// ---------------------------------------------------------------------------
__global__ __launch_bounds__(256)
void edge_logit_max(const long long* __restrict__ srcI,
                    const long long* __restrict__ dstI, int E, int N,
                    const float* __restrict__ es, const float* __restrict__ ed,
                    float* __restrict__ ews, float* __restrict__ m) {
  const int i = blockIdx.x * blockDim.x + threadIdx.x;
  const int ET = E + N;
  if (i >= ET) return;
  int s_, d_;
  if (i < E) { s_ = (int)srcI[i]; d_ = (int)dstI[i]; }
  else       { s_ = d_ = i - E; }
  float v = es[s_] + ed[d_];
  v = (v > 0.f) ? v : NEG_SLOPE * v;
  ews[i] = v;
  atomicMaxF(m + d_, v);
}

// ---------------------------------------------------------------------------
// Pass 2: ex = exp(e - m[dst]); store in place; segment sum to s.
// ---------------------------------------------------------------------------
__global__ __launch_bounds__(256)
void edge_exp_sum(const long long* __restrict__ dstI, int E, int N,
                  float* __restrict__ ews, const float* __restrict__ m,
                  float* __restrict__ s) {
  const int i = blockIdx.x * blockDim.x + threadIdx.x;
  const int ET = E + N;
  if (i >= ET) return;
  const int d_ = (i < E) ? (int)dstI[i] : (i - E);
  const float ex = __expf(ews[i] - m[d_]);
  ews[i] = ex;
  atomicAddF(s + d_, ex);
}

// ---------------------------------------------------------------------------
// Pass 3: one wave per edge. alpha = ex / s[dst];
// accum[dst] += alpha * h[src]   (2 floats per lane, 64-wide).
// ---------------------------------------------------------------------------
__global__ __launch_bounds__(256)
void edge_aggregate(const long long* __restrict__ srcI,
                    const long long* __restrict__ dstI, int E, int N,
                    const float* __restrict__ ews, const float* __restrict__ s,
                    const float* __restrict__ H, float* __restrict__ accum) {
  const int wave = (int)((blockIdx.x * 256u + threadIdx.x) >> 5);
  const int lane = threadIdx.x & 31;
  const int ET = E + N;
  if (wave >= ET) return;
  int s_, d_;
  if (wave < E) { s_ = (int)srcI[wave]; d_ = (int)dstI[wave]; }
  else          { s_ = d_ = wave - E; }
  const float alpha = ews[wave] / s[d_];
  const float2 hv = *(const float2*)(H + (size_t)s_ * HID + lane * 2);
  float* acc = accum + (size_t)d_ * HID + lane * 2;
  atomicAddF(acc + 0, alpha * hv.x);
  atomicAddF(acc + 1, alpha * hv.y);
}

// ---------------------------------------------------------------------------
// out = accum + bias, optional ReLU. float4-vectorized.
// ---------------------------------------------------------------------------
__global__ __launch_bounds__(256)
void bias_act(const float* __restrict__ accum, const float* __restrict__ b,
              float* __restrict__ out, long long total4, int do_relu) {
  const long long idx = (long long)blockIdx.x * blockDim.x + threadIdx.x;
  if (idx >= total4) return;
  const float4 a = ((const float4*)accum)[idx];
  const float4 bb = ((const float4*)b)[idx & (HID / 4 - 1)];
  float4 v = {a.x + bb.x, a.y + bb.y, a.z + bb.z, a.w + bb.w};
  if (do_relu) {
    v.x = fmaxf(v.x, 0.f); v.y = fmaxf(v.y, 0.f);
    v.z = fmaxf(v.z, 0.f); v.w = fmaxf(v.w, 0.f);
  }
  ((float4*)out)[idx] = v;
}

// ---------------------------------------------------------------------------
extern "C" void kernel_launch(void* const* d_in, const int* in_sizes, int n_in,
                              void* d_out, int out_size, void* d_ws, size_t ws_size,
                              hipStream_t stream) {
  const float*     x  = (const float*)d_in[0];
  const long long* ei = (const long long*)d_in[1];
  const int N = in_sizes[0] / 128;
  const int E = in_sizes[1] / 2;
  const long long* srcI = ei;
  const long long* dstI = ei + E;
  const int ET = E + N;

  // workspace carve-up (floats)
  float* ws    = (float*)d_ws;
  float* h     = ws;  ws += (size_t)N * HID;   // GEMM output
  float* accum = ws;  ws += (size_t)N * HID;   // aggregation accumulator
  float* feat  = ws;  ws += (size_t)N * HID;   // layer output / next input
  float* es    = ws;  ws += N;
  float* ed    = ws;  ws += N;
  float* mbuf  = ws;  ws += N;
  float* sbuf  = ws;  ws += N;
  float* ews   = ws;                           // per-edge logit -> exp

  const float* Wl[3]  = {(const float*)d_in[2],  (const float*)d_in[6],  (const float*)d_in[10]};
  const float* asr[3] = {(const float*)d_in[3],  (const float*)d_in[7],  (const float*)d_in[11]};
  const float* ads[3] = {(const float*)d_in[4],  (const float*)d_in[8],  (const float*)d_in[12]};
  const float* bia[3] = {(const float*)d_in[5],  (const float*)d_in[9],  (const float*)d_in[13]};
  const int Kdim[3]   = {128, HID, HID};

  const dim3 blk(256);
  const int rowTiles = (N + 15) / 16;
  const dim3 gGemm((rowTiles + 7) / 8);
  const dim3 gNode((N + 255) / 256);
  const dim3 gEdgeT((ET + 255) / 256);
  const dim3 gEdgeW((unsigned)((ET + 7) / 8));          // 8 waves / block
  const long long total4 = (long long)N * (HID / 4);
  const dim3 gBias((unsigned)((total4 + 255) / 256));

  const float* cur_in = x;
  for (int l = 0; l < 3; ++l) {
    gat_gemm_wmma<<<gGemm, blk, 0, stream>>>(cur_in, Wl[l], h, N, Kdim[l]);
    node_prep<<<gNode, blk, 0, stream>>>(h, asr[l], ads[l], es, ed, mbuf, sbuf, accum, N);
    edge_logit_max<<<gEdgeT, blk, 0, stream>>>(srcI, dstI, E, N, es, ed, ews, mbuf);
    edge_exp_sum<<<gEdgeT, blk, 0, stream>>>(dstI, E, N, ews, mbuf, sbuf);
    edge_aggregate<<<gEdgeW, blk, 0, stream>>>(srcI, dstI, E, N, ews, sbuf, h, accum);
    float* dstbuf = (l == 2) ? (float*)d_out : feat;
    bias_act<<<gBias, blk, 0, stream>>>(accum, bia[l], dstbuf, total4, (l < 2) ? 1 : 0);
    cur_in = feat;
  }
}
